// GCAFA_39943195853379
// MI455X (gfx1250) — matrix operand order repaired and checked by reading, not compiled, not run on hardware
//
#include <hip/hip_runtime.h>
#include <hip/hip_bf16.h>
#include <stddef.h>

// GCAFA axial-attention block for MI455X (gfx1250, wave32, WMMA).
// All GEMMs on v_wmma_f32_16x16x32_bf16. Weights pre-converted to bf16 so
// A-fragments are two b128 global loads (no per-tile cvt/perm VALU).
// q/k and attention output stored pre-transposed [w][c] so all bf16->LDS
// staging is straight padded-row copies done with GLOBAL_LOAD_ASYNC_TO_LDS_B128
// (ASYNCcnt), overlapping the softmax. BN folded; PReLU+residual in epilogues.

#define Bn   8
#define Cch  128
#define Hh   224
#define Ww   224
#define CAtt 64
#define HWp  (Hh * Ww)

typedef __attribute__((ext_vector_type(16))) __bf16       v16bf;
typedef __attribute__((ext_vector_type(8)))  float        v8f;
typedef __attribute__((ext_vector_type(8)))  unsigned int v8u;

// 16-byte int vector type matching the async-builtin's expected pointee
// (diagnostic: "__attribute__((__vector_size__(4 * sizeof(int)))) int")
typedef int v4i __attribute__((vector_size(16)));
typedef __attribute__((address_space(1))) v4i gv4i;  // global
typedef __attribute__((address_space(3))) v4i lv4i;  // LDS

__device__ __forceinline__ unsigned short bfbits(float f) {
  __bf16 h = (__bf16)f;
  return __builtin_bit_cast(unsigned short, h);
}
__device__ __forceinline__ unsigned pack2(float lo, float hi) {
  return ((unsigned)bfbits(hi) << 16) | (unsigned)bfbits(lo);
}
__device__ __forceinline__ float preluf(float x, float a) {
  return x >= 0.f ? x : a * x;
}

// ---- async global->LDS copy (16B per lane), with safe fallback -------------
__device__ __forceinline__ void async_cp16(const unsigned short* gsrc,
                                           unsigned short* ldst) {
#if __has_builtin(__builtin_amdgcn_global_load_async_to_lds_b128)
  __builtin_amdgcn_global_load_async_to_lds_b128((gv4i*)gsrc, (lv4i*)ldst, 0, 0);
#else
  *(uint4*)ldst = *(const uint4*)gsrc;
#endif
}
__device__ __forceinline__ void wait_async0() {
#if __has_builtin(__builtin_amdgcn_s_wait_asynccnt)
  __builtin_amdgcn_s_wait_asynccnt(0);
#elif __has_builtin(__builtin_amdgcn_global_load_async_to_lds_b128)
  asm volatile("s_wait_asynccnt 0x0" ::: "memory");
#endif
}

// ---- WMMA fragment loaders (ISA 7.12.2 index maps) -------------------------
// A (16-bit 16x32): lane half g, VGPR r: K = (r<4 ? 2r : 16+2(r-4)) + 8g,
// packed pairs -> two contiguous 16B chunks at kbase+8g and kbase+16+8g.
__device__ __forceinline__ v16bf fragA(const unsigned short* rowp, int kbase,
                                       int g) {
  uint4 lo = *(const uint4*)(rowp + kbase + 8 * g);
  uint4 hi = *(const uint4*)(rowp + kbase + 16 + 8 * g);
  v8u u;
  u[0] = lo.x; u[1] = lo.y; u[2] = lo.z; u[3] = lo.w;
  u[4] = hi.x; u[5] = hi.y; u[6] = hi.z; u[7] = hi.w;
  return __builtin_bit_cast(v16bf, u);
}
// B (16-bit 32x16): same K map as A -> two 16B chunks at kbase+8g and kbase+16+8g.
__device__ __forceinline__ v16bf fragB(const unsigned short* colp, int kbase,
                                       int g) {
  uint4 lo = *(const uint4*)(colp + kbase + 8 * g);
  uint4 hi = *(const uint4*)(colp + kbase + 16 + 8 * g);
  v8u u;
  u[0] = lo.x; u[1] = lo.y; u[2] = lo.z; u[3] = lo.w;
  u[4] = hi.x; u[5] = hi.y; u[6] = hi.z; u[7] = hi.w;
  return __builtin_bit_cast(v16bf, u);
}
__device__ __forceinline__ uint4 packrow(const v8f& a) {
  uint4 o;
  o.x = pack2(a[0], a[1]); o.y = pack2(a[2], a[3]);
  o.z = pack2(a[4], a[5]); o.w = pack2(a[6], a[7]);
  return o;
}

// ---------------------------------------------------------------- kernel 0a
// Fold BatchNorms: prm = sc1[0] sh1[192] sc2[384] sh2[512] sc3[640] sh3[768].
__global__ void bn_prep(const float* g1, const float* b1, const float* m1, const float* v1,
                        const float* g2, const float* b2, const float* m2, const float* v2,
                        const float* g3, const float* b3, const float* m3, const float* v3,
                        float* prm) {
  int i = threadIdx.x + blockIdx.x * blockDim.x;
  if (i < 192) {
    float s = g1[i] * rsqrtf(v1[i] + 1e-5f);
    prm[i] = s; prm[192 + i] = b1[i] - m1[i] * s;
  } else if (i < 320) {
    int c = i - 192;
    float s = g2[c] * rsqrtf(v2[c] + 1e-5f);
    prm[384 + c] = s; prm[512 + c] = b2[c] - m2[c] * s;
  } else if (i < 448) {
    int c = i - 320;
    float s = g3[c] * rsqrtf(v3[c] + 1e-5f);
    prm[640 + c] = s; prm[768 + c] = b3[c] - m3[c] * s;
  }
}

// ---------------------------------------------------------------- kernel 0b
// Pre-convert the three weight matrices to bf16 (row-major [oc][K]).
__global__ void wcvt(const float* wq, const float* wp, const float* wg,
                     unsigned short* q, unsigned short* p, unsigned short* g2) {
  int i = threadIdx.x + blockIdx.x * 256;
  if (i < 24576) q[i] = bfbits(wq[i]);
  else if (i < 32768) p[i - 24576] = bfbits(wp[i - 24576]);
  else if (i < 49152) g2[i - 32768] = bfbits(wg[i - 32768]);
}

// ---------------------------------------------------------------- kernel 1
// f_qkv = prelu(bn1(Wqkv @ x)) : per (b,h) GEMM M=192 K=128 N=224.
// ws1 per (b,h): qT[224][64] | kT[224][64] | v[64][224]   (bf16, q pre-scaled)
__global__ __launch_bounds__(256) void qkv_kernel(
    const float* __restrict__ x, const unsigned short* __restrict__ wq,
    const float* __restrict__ prm, const float* __restrict__ a1p,
    unsigned short* __restrict__ ws1) {
  __shared__ __align__(16) unsigned int Xs[Ww * 68];  // [w][c] bf16 pairs, pad 68
  const unsigned short* Xsu = (const unsigned short*)Xs;
  const int bh = blockIdx.x;
  const int b = bh / Hh, h = bh - b * Hh;
  const float* xrow = x + (size_t)b * Cch * HWp + (size_t)h * Ww;
  for (int idx = threadIdx.x; idx < 64 * Ww; idx += 256) {
    int cp = idx / Ww, w = idx - cp * Ww;  // channel pair, coalesced over w
    float f0 = xrow[(size_t)(2 * cp) * HWp + w];
    float f1 = xrow[(size_t)(2 * cp + 1) * HWp + w];
    Xs[w * 68 + cp] = pack2(f0, f1);
  }
  __syncthreads();
  const int lane = threadIdx.x & 31, wid = threadIdx.x >> 5;
  const int g = lane >> 4, l16 = lane & 15;
  const float slope = a1p[0];
  unsigned short* wsbh = ws1 + (size_t)bh * 3 * Ww * 64;
  for (int t = wid; t < 12 * 14; t += 8) {
    const int Mt = t / 14, Nt = t - Mt * 14;
    const unsigned short* wrow = wq + (size_t)(Mt * 16 + l16) * Cch;
    const unsigned short* bcol = Xsu + (Nt * 16 + l16) * 136;
    v8f acc = {};
#pragma unroll
    for (int ks = 0; ks < 4; ++ks)
      acc = __builtin_amdgcn_wmma_f32_16x16x32_bf16(
          false, fragA(wrow, 32 * ks, g), false, fragB(bcol, 32 * ks, g),
          (short)0, acc, false, false);
    const int w = Nt * 16 + l16;
    v8f vv;
#pragma unroll
    for (int r = 0; r < 8; ++r) {
      const int oc = Mt * 16 + r + 8 * g;
      float f = preluf(acc[r] * prm[oc] + prm[192 + oc], slope);
      if (oc < CAtt) f *= 0.125f;  // fold 1/sqrt(CA) into q
      vv[r] = f;
    }
    if (Mt < 4) {          // q -> transposed [w][64]
      *(uint4*)(wsbh + w * 64 + Mt * 16 + 8 * g) = packrow(vv);
    } else if (Mt < 8) {   // k -> transposed [w][64]
      *(uint4*)(wsbh + Ww * 64 + w * 64 + (Mt - 4) * 16 + 8 * g) = packrow(vv);
    } else {               // v -> natural [64][224]
#pragma unroll
      for (int r = 0; r < 8; ++r)
        wsbh[2 * Ww * 64 + (Mt * 16 + r + 8 * g - 128) * Ww + w] = bfbits(vv[r]);
    }
  }
}

// ---------------------------------------------------------------- kernel 2
// Axial attention for one (b,h, 32-query tile).
// Phase 1: S = q^T k (M=query K=c=64 N=key=224), accum in regs.
// Softmax via half-wave shfl_xor (C layout M=r+8g, N=lane%16).
// Phase 2: O = V P^T (M=c K=key N=query). Output ws2[B,H,224,64] bf16 (b128).
__global__ __launch_bounds__(64) void attn_kernel(
    const unsigned short* __restrict__ qkv, unsigned short* __restrict__ att) {
  __shared__ __align__(16) unsigned short smem[22272];  // 44.5 KB, overlaid
  unsigned short* qT = smem;          // phase1: [32][72]
  unsigned short* kT = smem + 2304;   // phase1: [224][72]
  unsigned short* Sb = smem;          // phase2: [32][232]
  unsigned short* Vs = smem + 7424;   // phase2: [64][232]
  const int blk = blockIdx.x;
  const int qt = blk % 7, bh = blk / 7;
  const unsigned short* qTg = qkv + (size_t)bh * 3 * Ww * 64;  // [224][64]
  const unsigned short* kTg = qTg + Ww * 64;                   // [224][64]
  const unsigned short* vg  = qTg + 2 * Ww * 64;               // [64][224]
  // async-stage qT (32x64) and kT (224x64), padded to stride 72
  for (int i = threadIdx.x; i < 32 * 8; i += 64) {
    int row = i >> 3, ch = i & 7;
    async_cp16(qTg + (qt * 32 + row) * 64 + ch * 8, qT + row * 72 + ch * 8);
  }
  for (int i = threadIdx.x; i < 224 * 8; i += 64) {
    int row = i >> 3, ch = i & 7;
    async_cp16(kTg + row * 64 + ch * 8, kT + row * 72 + ch * 8);
  }
  wait_async0();
  __syncthreads();
  const int lane = threadIdx.x & 31, wid = threadIdx.x >> 5;
  const int g = lane >> 4, l16 = lane & 15;
  v8f S[14];
  const v8f vzero = {};
#pragma unroll
  for (int n = 0; n < 14; ++n) S[n] = vzero;
  const unsigned short* arow = qT + (wid * 16 + l16) * 72;
#pragma unroll
  for (int ks = 0; ks < 2; ++ks) {
    v16bf av = fragA(arow, 32 * ks, g);
#pragma unroll
    for (int n = 0; n < 14; ++n)
      S[n] = __builtin_amdgcn_wmma_f32_16x16x32_bf16(
          false, av, false, fragB(kT + (n * 16 + l16) * 72, 32 * ks, g),
          (short)0, S[n], false, false);
  }
  __syncthreads();  // all waves done with qT/kT before region reuse
  // async-stage V (overlaps softmax below)
  for (int i = threadIdx.x; i < 64 * 28; i += 64) {
    int row = i / 28, ch = i - row * 28;
    async_cp16(vg + row * Ww + ch * 8, Vs + row * 232 + ch * 8);
  }
  float inv[8];
#pragma unroll
  for (int r = 0; r < 8; ++r) {
    float m = -3.0e38f;
#pragma unroll
    for (int n = 0; n < 14; ++n) m = fmaxf(m, S[n][r]);
#pragma unroll
    for (int msk = 1; msk < 16; msk <<= 1) m = fmaxf(m, __shfl_xor(m, msk, 32));
    float s = 0.f;
#pragma unroll
    for (int n = 0; n < 14; ++n) { float e = __expf(S[n][r] - m); S[n][r] = e; s += e; }
#pragma unroll
    for (int msk = 1; msk < 16; msk <<= 1) s += __shfl_xor(s, msk, 32);
    inv[r] = 1.f / s;
  }
#pragma unroll
  for (int r = 0; r < 8; ++r) {
    const int mloc = wid * 16 + r + 8 * g;
#pragma unroll
    for (int n = 0; n < 14; ++n)
      Sb[mloc * 232 + n * 16 + l16] = bfbits(S[n][r] * inv[r]);
  }
  wait_async0();
  __syncthreads();
  // O = V @ P^T ; store transposed [w][64] with b128 stores
  unsigned short* ob = att + (size_t)bh * Ww * 64 + (qt * 32 + wid * 16 + l16) * 64;
  const unsigned short* bcol = Sb + (wid * 16 + l16) * 232;
#pragma unroll
  for (int Mt = 0; Mt < 4; ++Mt) {
    v8f acc = {};
    const unsigned short* arow2 = Vs + (Mt * 16 + l16) * 232;
#pragma unroll
    for (int ks = 0; ks < 7; ++ks)
      acc = __builtin_amdgcn_wmma_f32_16x16x32_bf16(
          false, fragA(arow2, 32 * ks, g), false, fragB(bcol, 32 * ks, g),
          (short)0, acc, false, false);
    *(uint4*)(ob + Mt * 16 + 8 * g) = packrow(acc);
  }
}

// ---------------------------------------------------------------- kernel 3
// out1 = prelu(bn2(Wproj @ f_att)) + input : per (b,h) GEMM M=128 K=64 N=224.
__global__ __launch_bounds__(256) void proj_kernel(
    const unsigned short* __restrict__ att, const unsigned short* __restrict__ wp,
    const float* __restrict__ prm, const float* __restrict__ a2p,
    const float* __restrict__ x, float* __restrict__ out1) {
  __shared__ __align__(16) unsigned short Bs[Ww * 72];  // [w][c] pad 72
  const int bh = blockIdx.x;
  const int b = bh / Hh, h = bh - b * Hh;
  const unsigned short* ar = att + (size_t)bh * Ww * 64;  // [224][64]
  for (int i = threadIdx.x; i < 224 * 8; i += 256) {
    int row = i >> 3, ch = i & 7;
    async_cp16(ar + row * 64 + ch * 8, Bs + row * 72 + ch * 8);
  }
  wait_async0();
  __syncthreads();
  const int lane = threadIdx.x & 31, wid = threadIdx.x >> 5;
  const int g = lane >> 4, l16 = lane & 15;
  const float slope = a2p[0];
  for (int t = wid; t < 8 * 14; t += 8) {
    const int Mt = t / 14, Nt = t - Mt * 14;
    const unsigned short* wrow = wp + (size_t)(Mt * 16 + l16) * 64;
    const unsigned short* bcol = Bs + (Nt * 16 + l16) * 72;
    v8f acc = {};
#pragma unroll
    for (int ks = 0; ks < 2; ++ks)
      acc = __builtin_amdgcn_wmma_f32_16x16x32_bf16(
          false, fragA(wrow, 32 * ks, g), false, fragB(bcol, 32 * ks, g),
          (short)0, acc, false, false);
    const int w = Nt * 16 + l16;
#pragma unroll
    for (int r = 0; r < 8; ++r) {
      const int oc = Mt * 16 + r + 8 * g;
      float vv = preluf(acc[r] * prm[384 + oc] + prm[512 + oc], slope);
      const size_t gi = ((size_t)b * Cch + oc) * HWp + (size_t)h * Ww + w;
      out1[gi] = vv + x[gi];
    }
  }
}

// ---------------------------------------------------------------- kernel 4
// out = prelu(bn3(Wg2 @ out1)) + out1 : per (b,h) GEMM M=128 K=128 N=224.
__global__ __launch_bounds__(256) void conv2_kernel(
    const float* __restrict__ out1, const unsigned short* __restrict__ wg,
    const float* __restrict__ prm, const float* __restrict__ a3p,
    float* __restrict__ out) {
  __shared__ __align__(16) unsigned int Xs[Ww * 68];  // [w][c] pairs, pad 68
  const unsigned short* Xsu = (const unsigned short*)Xs;
  const int bh = blockIdx.x;
  const int b = bh / Hh, h = bh - b * Hh;
  const float* xr = out1 + (size_t)b * Cch * HWp + (size_t)h * Ww;
  for (int idx = threadIdx.x; idx < 64 * Ww; idx += 256) {
    int cp = idx / Ww, w = idx - cp * Ww;
    float f0 = xr[(size_t)(2 * cp) * HWp + w];
    float f1 = xr[(size_t)(2 * cp + 1) * HWp + w];
    Xs[w * 68 + cp] = pack2(f0, f1);
  }
  __syncthreads();
  const int lane = threadIdx.x & 31, wid = threadIdx.x >> 5;
  const int g = lane >> 4, l16 = lane & 15;
  const float slope = a3p[0];
  for (int t = wid; t < 8 * 14; t += 8) {
    const int Mt = t / 14, Nt = t - Mt * 14;
    const unsigned short* wrow = wg + (size_t)(Mt * 16 + l16) * Cch;
    const unsigned short* bcol = Xsu + (Nt * 16 + l16) * 136;
    v8f acc = {};
#pragma unroll
    for (int ks = 0; ks < 4; ++ks)
      acc = __builtin_amdgcn_wmma_f32_16x16x32_bf16(
          false, fragA(wrow, 32 * ks, g), false, fragB(bcol, 32 * ks, g),
          (short)0, acc, false, false);
    const int w = Nt * 16 + l16;
#pragma unroll
    for (int r = 0; r < 8; ++r) {
      const int oc = Mt * 16 + r + 8 * g;
      float vv = preluf(acc[r] * prm[640 + oc] + prm[768 + oc], slope);
      const size_t gi = ((size_t)b * Cch + oc) * HWp + (size_t)h * Ww + w;
      out[gi] = vv + xr[(size_t)oc * HWp + w];
    }
  }
}

// ---------------------------------------------------------------- launch
extern "C" void kernel_launch(void* const* d_in, const int* in_sizes, int n_in,
                              void* d_out, int out_size, void* d_ws, size_t ws_size,
                              hipStream_t stream) {
  const float* x     = (const float*)d_in[0];
  const float* wqkv  = (const float*)d_in[1];
  const float* bn1g  = (const float*)d_in[2];
  const float* bn1b  = (const float*)d_in[3];
  const float* bn1m  = (const float*)d_in[4];
  const float* bn1v  = (const float*)d_in[5];
  const float* a1p   = (const float*)d_in[6];
  const float* wproj = (const float*)d_in[7];
  const float* bn2g  = (const float*)d_in[8];
  const float* bn2b  = (const float*)d_in[9];
  const float* bn2m  = (const float*)d_in[10];
  const float* bn2v  = (const float*)d_in[11];
  const float* a2p   = (const float*)d_in[12];
  const float* wg2   = (const float*)d_in[13];
  const float* bn3g  = (const float*)d_in[14];
  const float* bn3b  = (const float*)d_in[15];
  const float* bn3m  = (const float*)d_in[16];
  const float* bn3v  = (const float*)d_in[17];
  const float* a3p   = (const float*)d_in[18];

  // workspace: [prm|bf16 weights 128KB][qkv bf16 154MB][attn bf16 51MB][out1 f32 205MB]
  char* ws = (char*)d_ws;
  float*          prm   = (float*)ws;
  unsigned short* wq_bf = (unsigned short*)(ws + 4096);
  unsigned short* wp_bf = (unsigned short*)(ws + 53248);
  unsigned short* wg_bf = (unsigned short*)(ws + 69632);
  unsigned short* ws1   = (unsigned short*)(ws + 131072);
  unsigned short* ws2   = (unsigned short*)(ws + 131072 + (size_t)154140672);
  float*          out1  = (float*)(ws + 131072 + (size_t)154140672 + (size_t)51380224);

  bn_prep<<<2, 256, 0, stream>>>(bn1g, bn1b, bn1m, bn1v,
                                 bn2g, bn2b, bn2m, bn2v,
                                 bn3g, bn3b, bn3m, bn3v, prm);
  wcvt<<<192, 256, 0, stream>>>(wqkv, wproj, wg2, wq_bf, wp_bf, wg_bf);
  qkv_kernel<<<Bn * Hh, 256, 0, stream>>>(x, wq_bf, prm, a1p, ws1);
  attn_kernel<<<Bn * Hh * 7, 64, 0, stream>>>(ws1, ws2);
  proj_kernel<<<Bn * Hh, 256, 0, stream>>>(ws2, wp_bf, prm, a2p, x, out1);
  conv2_kernel<<<Bn * Hh, 256, 0, stream>>>(out1, wg_bf, prm, a3p, (float*)d_out);
}
